// SimpleRNN_49692771615219
// MI455X (gfx1250) — matrix operand order repaired and checked by reading, not compiled, not run on hardware
//
#include <hip/hip_runtime.h>

// -------- CDNA5 WMMA types --------
typedef __attribute__((ext_vector_type(16))) __bf16 v16bf;
typedef __attribute__((ext_vector_type(8)))  float  v8f;

#define RNN_B 256
#define RNN_T 512
#define RNN_D 64
#define RNN_H 256
#define RNN_OUT 24

// Hardware tanh: gfx1250 has V_TANH_F32 (TRANS unit, co-executes with WMMA).
__device__ __forceinline__ float fast_tanh(float x) {
#if __has_builtin(__builtin_amdgcn_tanhf)
    return __builtin_amdgcn_tanhf(x);
#else
    float r;
    asm volatile("v_tanh_f32 %0, %1" : "=v"(r) : "v"(x));
    return r;
#endif
}

// =====================================================================
// Kernel 1: pre[M,256] = in[M,K] @ W[256,K]^T + (bih+bhh)
// bf16 WMMA with fp32 accumulate. Block = 256 thr (8 waves).
// Block tile: M=64 x N=256. Each wave: one 16-row M-subtile x 8 N-subtiles.
// =====================================================================
template <typename TIN>
__global__ void __launch_bounds__(256)
gemm_pre(const TIN* __restrict__ in,    // [M,K]
         const float* __restrict__ W,   // [256,K] fp32
         const float* __restrict__ bih,
         const float* __restrict__ bhh,
         float* __restrict__ out,       // [M,256]
         int M, int K)
{
    __shared__ __bf16 Ash[64][34];    // +2 pad vs LDS bank conflicts
    __shared__ __bf16 Bsh[256][34];

    const int tid  = threadIdx.x;
    const int wave = tid >> 5;
    const int lane = tid & 31;
    const int l16  = lane & 15;
    const int lhi  = lane >> 4;       // 0/1

    const int mrow0 = blockIdx.x * 64;
    const int msub  = wave >> 1;      // 0..3
    const int nbase = (wave & 1) * 8; // first N-subtile index (of 16)

    // init accumulators with bias (same bias for every M row of a column)
    v8f acc[8];
#pragma unroll
    for (int i = 0; i < 8; ++i) {
        const int col = (nbase + i) * 16 + l16;
        const float bv = bih[col] + bhh[col];
#pragma unroll
        for (int r = 0; r < 8; ++r) acc[i][r] = bv;
    }

    for (int k0 = 0; k0 < K; k0 += 32) {
        // stage A tile 64x32 (8 elems/thread), convert to bf16
#pragma unroll
        for (int i = 0; i < 8; ++i) {
            const int idx = tid + i * 256;
            const int r = idx >> 5, c = idx & 31;
            Ash[r][c] = (__bf16)(float)in[(size_t)(mrow0 + r) * K + k0 + c];
        }
        // stage B tile 256x32 (32 elems/thread) from fp32 W
#pragma unroll
        for (int i = 0; i < 32; ++i) {
            const int idx = tid + i * 256;
            const int r = idx >> 5, c = idx & 31;
            Bsh[r][c] = (__bf16)W[(size_t)r * K + k0 + c];
        }
        __syncthreads();

        // A fragment: 16x32 bf16. lane<16: row=lane, K {0..15}; lane>=16: K {16..31}
        v16bf a;
        {
            const int ar = msub * 16 + l16;
            const int kb = lhi * 16;
#pragma unroll
            for (int j = 0; j < 8; ++j) a[j]     = Ash[ar][kb + j];
#pragma unroll
            for (int j = 0; j < 8; ++j) a[j + 8] = Ash[ar][8 + kb + j];
        }
        // B fragments: 32x16 bf16. lane -> col, lhi selects K half; VGPR j holds K=kh+2j,kh+2j+1
#pragma unroll
        for (int i = 0; i < 8; ++i) {
            v16bf b;
            const int br = (nbase + i) * 16 + l16;
            const int kh = lhi * 16;
#pragma unroll
            for (int j = 0; j < 8; ++j) {
                b[2 * j]     = Bsh[br][kh + 2 * j];
                b[2 * j + 1] = Bsh[br][kh + 2 * j + 1];
            }
            acc[i] = __builtin_amdgcn_wmma_f32_16x16x32_bf16(
                false, a, false, b, (short)0, acc[i], false, false);
        }
        __syncthreads();
    }

    // store: C/D layout: VGPR r, lane -> M = r + 8*lhi, N = l16
    const int hi8 = lhi * 8;
#pragma unroll
    for (int i = 0; i < 8; ++i) {
        const int col = (nbase + i) * 16 + l16;
#pragma unroll
        for (int r = 0; r < 8; ++r) {
            out[(size_t)(mrow0 + msub * 16 + r + hi8) * RNN_H + col] = acc[i][r];
        }
    }
}

// =====================================================================
// Kernel 2: recurrent scan  h_t = tanh(pre_t + h_{t-1} @ Whh^T)
// One block per 16 batch rows (16 blocks). 8 waves; wave owns 2 N-subtiles.
// - Whh bf16 B-fragments register-resident for entire scan (128 VGPRs/wave)
// - double-buffered LDS h tile -> single barrier per timestep
// - pre_{t+1} loads software-pipelined under the WMMA block of step t
// - hardware V_TANH_F32
// =====================================================================
template <bool STORE_ALL>
__global__ void __launch_bounds__(256)
rnn_scan(const float* __restrict__ pre,  // [B,T,H]
         const float* __restrict__ Whh,  // [H,H] fp32
         __bf16* __restrict__ hout)      // STORE_ALL: [B,T,H] ; else [B,H]
{
    __shared__ __bf16 hbuf[2][16][264];  // h double buffer, padded

    const int tid  = threadIdx.x;
    const int wave = tid >> 5;
    const int lane = tid & 31;
    const int l16  = lane & 15;
    const int lhi  = lane >> 4;
    const int hi8  = lhi * 8;
    const int bbase = blockIdx.x * 16;

    // ---- preload time-invariant Whh B-fragments into registers ----
    // B[k][n] = Whh[n][k]; lane -> n within tile, lhi selects K half.
    v16bf bfr[2][8];
#pragma unroll
    for (int p = 0; p < 2; ++p) {
        const int n = (wave * 2 + p) * 16 + l16;
#pragma unroll
        for (int kc = 0; kc < 8; ++kc) {
            const float* wr = Whh + (size_t)n * RNN_H + kc * 32 + lhi * 16;
#pragma unroll
            for (int j = 0; j < 8; ++j) {
                bfr[p][kc][2 * j]     = (__bf16)wr[2 * j];
                bfr[p][kc][2 * j + 1] = (__bf16)wr[2 * j + 1];
            }
        }
    }

    // ---- h0 = 0 (only buffer 0 is read before being written) ----
    for (int i = tid; i < 16 * 264; i += 256)
        hbuf[0][i / 264][i % 264] = (__bf16)0.0f;

    // per-(p,r) base pointers into pre: row (bbase+r+hi8), column col; +t*H per step
    const int col0 = (wave * 2 + 0) * 16 + l16;
    const int col1 = (wave * 2 + 1) * 16 + l16;

    // ---- prologue: load pre_0 ----
    v8f cur[2];
#pragma unroll
    for (int r = 0; r < 8; ++r) {
        const size_t rowoff = (size_t)(bbase + r + hi8) * RNN_T * RNN_H;
        cur[0][r] = pre[rowoff + col0];
        cur[1][r] = pre[rowoff + col1];
    }
    __syncthreads();

    for (int t = 0; t < RNN_T; ++t) {
        // ---- issue loads for pre_{t+1} (overlaps with WMMA below) ----
        const int tn = (t + 1 < RNN_T) ? (t + 1) : (RNN_T - 1);
        v8f nxt[2];
#pragma unroll
        for (int r = 0; r < 8; ++r) {
            const size_t rowoff =
                (size_t)(bbase + r + hi8) * RNN_T * RNN_H + (size_t)tn * RNN_H;
            nxt[0][r] = pre[rowoff + col0];
            nxt[1][r] = pre[rowoff + col1];
            // pull pre tile for t+4 toward the caches
            __builtin_prefetch(&pre[rowoff + 3 * RNN_H + col0], 0, 1);
        }

        // ---- S = pre_t + h_{t-1} @ Whh^T ----
        const int rb = t & 1;          // read buffer
        v8f acc0 = cur[0], acc1 = cur[1];
#pragma unroll
        for (int kc = 0; kc < 8; ++kc) {
            v16bf a;
            const int kb = kc * 32 + lhi * 16;
#pragma unroll
            for (int j = 0; j < 8; ++j) a[j]     = hbuf[rb][l16][kb + j];
#pragma unroll
            for (int j = 0; j < 8; ++j) a[j + 8] = hbuf[rb][l16][kb + 8 + j];
            acc0 = __builtin_amdgcn_wmma_f32_16x16x32_bf16(
                false, a, false, bfr[0][kc], (short)0, acc0, false, false);
            acc1 = __builtin_amdgcn_wmma_f32_16x16x32_bf16(
                false, a, false, bfr[1][kc], (short)0, acc1, false, false);
        }

        // ---- h_t = tanh(S) -> write buffer, optional global store ----
        const int wb = (t + 1) & 1;    // write buffer
#pragma unroll
        for (int r = 0; r < 8; ++r) {
            const __bf16 h0b = (__bf16)fast_tanh(acc0[r]);
            const __bf16 h1b = (__bf16)fast_tanh(acc1[r]);
            const int m = r + hi8;
            hbuf[wb][m][col0] = h0b;
            hbuf[wb][m][col1] = h1b;
            if (STORE_ALL) {
                const size_t o = ((size_t)(bbase + m) * RNN_T + t) * RNN_H;
                hout[o + col0] = h0b;
                hout[o + col1] = h1b;
            }
        }
        __syncthreads();               // single barrier per step

        cur[0] = nxt[0];
        cur[1] = nxt[1];
    }

    if (!STORE_ALL) {
        // final hidden state h_{T-1} lives in buffer RNN_T & 1
        const int fb = RNN_T & 1;
#pragma unroll
        for (int r = 0; r < 8; ++r) {
            const int m = r + hi8;
            hout[(size_t)(bbase + m) * RNN_H + col0] = hbuf[fb][m][col0];
            hout[(size_t)(bbase + m) * RNN_H + col1] = hbuf[fb][m][col1];
        }
    }
}

// =====================================================================
// Kernel 3: out[b,o] = h2[b,:] . Wd[o,:] + bd[o]   (256 x 24, tiny)
// =====================================================================
__global__ void dense_out(const __bf16* __restrict__ h2,  // [B,H] bf16
                          const float* __restrict__ Wd,   // [24,H]
                          const float* __restrict__ bd,   // [24]
                          float* __restrict__ out)        // [B,24]
{
    const int idx = blockIdx.x * blockDim.x + threadIdx.x;
    if (idx >= RNN_B * RNN_OUT) return;
    const int b = idx / RNN_OUT;
    const int o = idx % RNN_OUT;
    float s = bd[o];
    for (int k = 0; k < RNN_H; ++k)
        s += (float)h2[(size_t)b * RNN_H + k] * Wd[(size_t)o * RNN_H + k];
    out[idx] = s;
}

// =====================================================================
// Launch
// inputs: 0 x, 1 Wih1, 2 Whh1, 3 bih1, 4 bhh1, 5 Wih2, 6 Whh2, 7 bih2,
//         8 bhh2, 9 Wd, 10 bd
// =====================================================================
extern "C" void kernel_launch(void* const* d_in, const int* in_sizes, int n_in,
                              void* d_out, int out_size, void* d_ws, size_t ws_size,
                              hipStream_t stream) {
    const float* x    = (const float*)d_in[0];
    const float* Wih1 = (const float*)d_in[1];
    const float* Whh1 = (const float*)d_in[2];
    const float* bih1 = (const float*)d_in[3];
    const float* bhh1 = (const float*)d_in[4];
    const float* Wih2 = (const float*)d_in[5];
    const float* Whh2 = (const float*)d_in[6];
    const float* bih2 = (const float*)d_in[7];
    const float* bhh2 = (const float*)d_in[8];
    const float* Wd   = (const float*)d_in[9];
    const float* bd   = (const float*)d_in[10];
    float* out = (float*)d_out;

    const size_t MT = (size_t)RNN_B * RNN_T;           // 131072 rows
    float*  pre = (float*)d_ws;                        // [B,T,H] fp32 (reused)
    __bf16* h1  = (__bf16*)((char*)d_ws + MT * RNN_H * sizeof(float));
    __bf16* h2  = h1 + MT * RNN_H;                     // [B,H] bf16

    const int gemm_blocks = (int)(MT / 64);            // 2048

    // layer 1: pre1 = x @ Wih1^T + b
    gemm_pre<float><<<gemm_blocks, 256, 0, stream>>>(x, Wih1, bih1, bhh1,
                                                     pre, (int)MT, RNN_D);
    // layer 1 scan -> h1 (bf16, all timesteps)
    rnn_scan<true><<<RNN_B / 16, 256, 0, stream>>>(pre, Whh1, h1);
    // layer 2: pre2 = h1 @ Wih2^T + b   (reuse pre buffer)
    gemm_pre<__bf16><<<gemm_blocks, 256, 0, stream>>>(h1, Wih2, bih2, bhh2,
                                                      pre, (int)MT, RNN_H);
    // layer 2 scan -> h2 (last timestep only)
    rnn_scan<false><<<RNN_B / 16, 256, 0, stream>>>(pre, Whh2, h2);
    // dense head
    dense_out<<<(RNN_B * RNN_OUT + 255) / 256, 256, 0, stream>>>(h2, Wd, bd, out);
}
